// BitLinearTriton_53068615910183
// MI455X (gfx1250) — 8-bit matrix operand order repaired and checked by reading, compiled, not run on hardware
//
#include <hip/hip_runtime.h>
#include <hip/hip_bf16.h>
#include <cstdint>
#include <cstddef>

typedef __attribute__((ext_vector_type(8))) int v8i;
typedef __attribute__((ext_vector_type(4))) int v4i;
typedef __attribute__((ext_vector_type(2))) int v2i;

#define K_DIM 2048
#define M_TOT 8192
#define N_TOT 8192
#define EPSF  1e-5f
#define W_ELEMS (N_TOT * K_DIM)
#define W_F4    (W_ELEMS / 4)

__global__ void k_wabs_partial(const float* __restrict__ w,
                               float* __restrict__ partial) {
  const int tid = threadIdx.x;
  const float4* w4 = reinterpret_cast<const float4*>(w);
  float s = 0.0f;
  for (int i = blockIdx.x * 256 + tid; i < W_F4; i += 512 * 256) {
    float4 v = w4[i];
    s += fabsf(v.x) + fabsf(v.y) + fabsf(v.z) + fabsf(v.w);
  }
  __shared__ float red[256];
  red[tid] = s;
  __syncthreads();
  for (int off = 128; off > 0; off >>= 1) {
    if (tid < off) red[tid] += red[tid + off];
    __syncthreads();
  }
  if (tid == 0) partial[blockIdx.x] = red[0];
}

__global__ void k_wgamma(const float* __restrict__ partial,
                         float* __restrict__ gamma_out) {
  __shared__ float red[512];
  const int tid = threadIdx.x;
  red[tid] = partial[tid];
  __syncthreads();
  for (int off = 256; off > 0; off >>= 1) {
    if (tid < off) red[tid] += red[tid + off];
    __syncthreads();
  }
  if (tid == 0) gamma_out[0] = fmaxf(red[0] * (1.0f / (float)W_ELEMS), EPSF);
}

__global__ void k_wquant(const float* __restrict__ w,
                         const float* __restrict__ gamma_p,
                         int* __restrict__ wq_packed) {
  const float inv = 1.0f / gamma_p[0];
  const int i = blockIdx.x * 256 + threadIdx.x;
  float4 v = reinterpret_cast<const float4*>(w)[i];
  int q0 = (int)rintf(fminf(fmaxf(v.x * inv, -1.0f), 1.0f));
  int q1 = (int)rintf(fminf(fmaxf(v.y * inv, -1.0f), 1.0f));
  int q2 = (int)rintf(fminf(fmaxf(v.z * inv, -1.0f), 1.0f));
  int q3 = (int)rintf(fminf(fmaxf(v.w * inv, -1.0f), 1.0f));
  wq_packed[i] = (q0 & 255) | ((q1 & 255) << 8) | ((q2 & 255) << 16) | ((q3 & 255) << 24);
}

__global__ void k_lnq(const float* __restrict__ x,
                      int* __restrict__ xq_packed,
                      float* __restrict__ xscale) {
  const int row = blockIdx.x;
  const int tid = threadIdx.x;
  const float4* xr = reinterpret_cast<const float4*>(x + (size_t)row * K_DIM);
  float4 a = xr[tid];
  float4 b = xr[tid + 256];

  float s  = a.x + a.y + a.z + a.w + b.x + b.y + b.z + b.w;
  float ss = a.x*a.x + a.y*a.y + a.z*a.z + a.w*a.w
           + b.x*b.x + b.y*b.y + b.z*b.z + b.w*b.w;

  __shared__ float r1[256];
  __shared__ float r2[256];
  r1[tid] = s; r2[tid] = ss;
  __syncthreads();
  for (int off = 128; off > 0; off >>= 1) {
    if (tid < off) { r1[tid] += r1[tid + off]; r2[tid] += r2[tid + off]; }
    __syncthreads();
  }
  const float mean = r1[0] * (1.0f / K_DIM);
  const float var  = r2[0] * (1.0f / K_DIM) - mean * mean;
  const float rstd = rsqrtf(var + EPSF);
  __syncthreads();

  float mx = 0.0f;
  mx = fmaxf(mx, fabsf(a.x - mean)); mx = fmaxf(mx, fabsf(a.y - mean));
  mx = fmaxf(mx, fabsf(a.z - mean)); mx = fmaxf(mx, fabsf(a.w - mean));
  mx = fmaxf(mx, fabsf(b.x - mean)); mx = fmaxf(mx, fabsf(b.y - mean));
  mx = fmaxf(mx, fabsf(b.z - mean)); mx = fmaxf(mx, fabsf(b.w - mean));
  r1[tid] = mx;
  __syncthreads();
  for (int off = 128; off > 0; off >>= 1) {
    if (tid < off) r1[tid] = fmaxf(r1[tid], r1[tid + off]);
    __syncthreads();
  }
  const float eta = fmaxf(r1[0] * rstd, EPSF);
  const float qs  = 127.0f / eta;

  auto quant4 = [&](float4 v) -> int {
    int q0 = (int)fminf(fmaxf(rintf((v.x - mean) * rstd * qs), -128.0f), 127.0f);
    int q1 = (int)fminf(fmaxf(rintf((v.y - mean) * rstd * qs), -128.0f), 127.0f);
    int q2 = (int)fminf(fmaxf(rintf((v.z - mean) * rstd * qs), -128.0f), 127.0f);
    int q3 = (int)fminf(fmaxf(rintf((v.w - mean) * rstd * qs), -128.0f), 127.0f);
    return (q0 & 255) | ((q1 & 255) << 8) | ((q2 & 255) << 16) | ((q3 & 255) << 24);
  };
  int* orow = xq_packed + (size_t)row * (K_DIM / 4);
  orow[tid]       = quant4(a);
  orow[tid + 256] = quant4(b);
  if (tid == 0) xscale[row] = eta * (1.0f / 127.0f);
}

__global__ void __launch_bounds__(256, 1)
k_gemm(const signed char* __restrict__ xq,
       const signed char* __restrict__ wq,
       const float* __restrict__ xscale,
       const float* __restrict__ gamma_p,
       const float* __restrict__ bias,
       float* __restrict__ out) {
  __shared__ __align__(16) signed char As[128 * 64];
  __shared__ __align__(16) signed char Bs[256 * 64];

  const int tid   = threadIdx.x;
  const int lane  = tid & 31;
  const int wave  = tid >> 5;
  const int lh    = lane & 15;
  const int hi    = lane >> 4;
  const int waveM = wave >> 2;
  const int waveN = wave & 3;

  const int m0 = blockIdx.y * 128;
  const int n0 = blockIdx.x * 256;

  v8i acc[4][4] = {};

  for (int k0 = 0; k0 < K_DIM; k0 += 64) {
#pragma unroll
    for (int i = 0; i < 2; ++i) {
      int idx = i * 256 + tid;
      int r = idx >> 2, c = (idx & 3) << 4;
      *reinterpret_cast<v4i*>(As + r * 64 + c) =
          *reinterpret_cast<const v4i*>(xq + (size_t)(m0 + r) * K_DIM + k0 + c);
    }
#pragma unroll
    for (int i = 0; i < 4; ++i) {
      int idx = i * 256 + tid;
      int r = idx >> 2, c = (idx & 3) << 4;
      *reinterpret_cast<v4i*>(Bs + r * 64 + c) =
          *reinterpret_cast<const v4i*>(wq + (size_t)(n0 + r) * K_DIM + k0 + c);
    }
    __syncthreads();

    v8i afrag[4];
    const int abase = hi * 16;
#pragma unroll
    for (int mi = 0; mi < 4; ++mi) {
      const signed char* rp = As + (waveM * 64 + mi * 16 + lh) * 64 + abase;
      v2i p0 = *reinterpret_cast<const v2i*>(rp);
      v2i p1 = *reinterpret_cast<const v2i*>(rp + 8);
      v2i p2 = *reinterpret_cast<const v2i*>(rp + 32);
      v2i p3 = *reinterpret_cast<const v2i*>(rp + 40);
      v8i t;
      t[0] = p0.x; t[1] = p0.y; t[2] = p1.x; t[3] = p1.y;
      t[4] = p2.x; t[5] = p2.y; t[6] = p3.x; t[7] = p3.y;
      afrag[mi] = t;
    }
    v8i bfrag[4];
    const int bbase = hi * 16;
#pragma unroll
    for (int ni = 0; ni < 4; ++ni) {
      const signed char* rp = Bs + (waveN * 64 + ni * 16 + lh) * 64 + bbase;
      v4i q0 = *reinterpret_cast<const v4i*>(rp);
      v4i q1 = *reinterpret_cast<const v4i*>(rp + 32);
      v8i t;
      t[0] = q0.x; t[1] = q0.y; t[2] = q0.z; t[3] = q0.w;
      t[4] = q1.x; t[5] = q1.y; t[6] = q1.z; t[7] = q1.w;
      bfrag[ni] = t;
    }

#pragma unroll
    for (int mi = 0; mi < 4; ++mi)
#pragma unroll
      for (int ni = 0; ni < 4; ++ni)
        acc[mi][ni] = __builtin_amdgcn_wmma_i32_16x16x64_iu8(
             true, afrag[mi],  true, bfrag[ni],
            acc[mi][ni],  false,  false);

    __syncthreads();
  }

  const float gamma = gamma_p[0];
#pragma unroll
  for (int mi = 0; mi < 4; ++mi) {
#pragma unroll
    for (int r = 0; r < 8; ++r) {
      const int m = m0 + waveM * 64 + mi * 16 + hi * 8 + r;
      const float rowscale = gamma * xscale[m];
#pragma unroll
      for (int ni = 0; ni < 4; ++ni) {
        const int n = n0 + waveN * 64 + ni * 16 + lh;
        out[(size_t)m * N_TOT + n] = (float)acc[mi][ni][r] * rowscale + bias[n];
      }
    }
  }
}

extern "C" void kernel_launch(void* const* d_in, const int* in_sizes, int n_in,
                              void* d_out, int out_size, void* d_ws, size_t ws_size,
                              hipStream_t stream) {
  (void)in_sizes; (void)n_in; (void)out_size; (void)ws_size;
  const float* x    = (const float*)d_in[0];
  const float* w    = (const float*)d_in[1];
  const float* bias = (const float*)d_in[2];
  float* out = (float*)d_out;

  char* ws = (char*)d_ws;
  float* g_gamma        = (float*)ws;
  float* g_part         = (float*)(ws + 256);
  float* g_xscale       = (float*)(ws + 4096);
  signed char* g_xq     = (signed char*)(ws + 65536);
  signed char* g_wq     = (signed char*)(ws + 65536 + (size_t)(16 << 20));

  k_wabs_partial<<<512, 256, 0, stream>>>(w, g_part);
  k_wgamma<<<1, 512, 0, stream>>>(g_part, g_gamma);
  k_wquant<<<W_F4 / 256, 256, 0, stream>>>(w, g_gamma, (int*)g_wq);
  k_lnq<<<M_TOT, 256, 0, stream>>>(x, (int*)g_xq, g_xscale);

  dim3 grid(N_TOT / 256, M_TOT / 128);
  k_gemm<<<grid, 256, 0, stream>>>(g_xq, g_wq, g_xscale, g_gamma, bias, out);
}
